// self_attention_mach_56788057587885
// MI455X (gfx1250) — matrix operand order repaired and checked by reading, not compiled, not run on hardware
//
#include <hip/hip_runtime.h>

// Fused self-attention for MI455X (gfx1250, wave32, WMMA 16x16x32 f16->f32).
// Phase 0: convert W -> f16 transposed (Wt[w][n][k]) for contiguous B-frags.
// Phase 1: Q/K/V = x @ W, f16 operands / f32 accumulate, 12 WMMA accs/wave.
// Phase 2: flash attention (online softmax, log2 domain); K/V tiles staged
//          into LDS with global_load_async_to_lds_b128 (ASYNCcnt path).
// Workspace: Qh(1MB) + Kh(1MB) + Vt(1MB) + Wt(192KB) ~= 3.2 MB.

typedef __attribute__((ext_vector_type(16))) _Float16 v16h;
typedef __attribute__((ext_vector_type(8)))  _Float16 v8h;
typedef __attribute__((ext_vector_type(8)))  float    v8f;
typedef __attribute__((ext_vector_type(4)))  float    v4f;
typedef _Float16 half_t;

#define SEQ  8192
#define DIN  512
#define DOUT 64
#define P1_WAVES 4
#define P2_WAVES 4

__device__ __forceinline__ v8f zero_v8f() {
    v8f z;
#pragma unroll
    for (int i = 0; i < 8; ++i) z[i] = 0.0f;
    return z;
}

__device__ __forceinline__ v16h combine(v8h lo, v8h hi) {
    return __builtin_shufflevector(lo, hi, 0, 1, 2, 3, 4, 5, 6, 7,
                                           8, 9, 10, 11, 12, 13, 14, 15);
}

// D = A(16x32 f16) * B(32x16 f16) + C(f32)
__device__ __forceinline__ v8f wmma_f16(v16h a, v16h b, v8f c) {
    return __builtin_amdgcn_wmma_f32_16x16x32_f16(
        /*neg_a=*/false, a, /*neg_b=*/false, b,
        /*c_mod=*/(short)0, c, /*reuse_a=*/false, /*reuse_b=*/false);
}

// ---------------------------------------------------------------------------
// Phase 0: Wt[w][n][k] = (f16) W_w[k][n]   (3 x 64 x 512)
// ---------------------------------------------------------------------------
__global__ __launch_bounds__(256) void wcvt_kernel(
    const float* __restrict__ Wq, const float* __restrict__ Wk,
    const float* __restrict__ Wv, half_t* __restrict__ Wt)
{
    const int idx = blockIdx.x * 256 + threadIdx.x;  // < 3*64*512
    const int w   = idx / (DOUT * DIN);
    const int rem = idx % (DOUT * DIN);
    const int n   = rem / DIN;
    const int k   = rem % DIN;
    const float* W = (w == 0) ? Wq : (w == 1) ? Wk : Wv;
    Wt[idx] = (half_t)W[k * DOUT + n];
}

// ---------------------------------------------------------------------------
// Phase 1: QKV projection. A-frag k order used here (contiguous, B matches): lane<16 -> row=lane,
// K in {0..15}; lane>=16 -> row=lane-16, K+16. B-frag: lane%16=col,
// (lane/16)*16 = K offset, contiguous K per lane -> two b128 loads from Wt.
// ---------------------------------------------------------------------------
__global__ __launch_bounds__(P1_WAVES * 32) void qkv_proj_kernel(
    const float* __restrict__ x, const half_t* __restrict__ Wt,
    half_t* __restrict__ Qh, half_t* __restrict__ Kh, half_t* __restrict__ Vt)
{
    const int lane    = threadIdx.x & 31;
    const int wave    = threadIdx.x >> 5;
    const int halfsel = lane >> 4;
    const int l16     = lane & 15;
    const int row0    = blockIdx.x * (P1_WAVES * 16) + wave * 16;

    v8f acc[3][4];
#pragma unroll
    for (int w = 0; w < 3; ++w)
#pragma unroll
        for (int jn = 0; jn < 4; ++jn) acc[w][jn] = zero_v8f();

    const float* xrow = x + (size_t)(row0 + l16) * DIN;

    for (int kk = 0; kk < DIN; kk += 32) {
        // A fragment from x: one contiguous 16-float run -> 4x b128 + cvt
        const float* ap = xrow + kk + 16 * halfsel;
        const v4f r0 = *(const v4f*)(ap);
        const v4f r1 = *(const v4f*)(ap + 4);
        const v4f r2 = *(const v4f*)(ap + 8);
        const v4f r3 = *(const v4f*)(ap + 12);
        v16h a;
#pragma unroll
        for (int j = 0; j < 4; ++j) {
            a[j]      = (half_t)r0[j];
            a[4 + j]  = (half_t)r1[j];
            a[8 + j]  = (half_t)r2[j];
            a[12 + j] = (half_t)r3[j];
        }
#pragma unroll
        for (int w = 0; w < 3; ++w) {
#pragma unroll
            for (int jn = 0; jn < 4; ++jn) {
                const half_t* wp = Wt + ((size_t)w * DOUT + jn * 16 + l16) * DIN
                                      + kk + 16 * halfsel;
                const v16h b = combine(*(const v8h*)wp, *(const v8h*)(wp + 8));
                acc[w][jn] = wmma_f16(a, b, acc[w][jn]);
            }
        }
    }

    // C/D layout: lane holds column n = jn*16 + lane%16, rows 8*(lane/16)+v.
#pragma unroll
    for (int jn = 0; jn < 4; ++jn) {
        v8h vq, vk, vv;
#pragma unroll
        for (int v = 0; v < 8; ++v) {
            vq[v] = (half_t)acc[0][jn][v];
            vk[v] = (half_t)acc[1][jn][v];
            vv[v] = (half_t)acc[2][jn][v];
        }
        const int c = jn * 16 + l16;
#pragma unroll
        for (int v = 0; v < 8; ++v) {
            const int r = row0 + 8 * halfsel + v;
            Qh[(size_t)r * DOUT + c] = vq[v];
            Kh[(size_t)r * DOUT + c] = vk[v];
        }
        // Vt row c, consecutive seq positions -> single 16B store
        *(v8h*)(Vt + (size_t)c * SEQ + row0 + 8 * halfsel) = vv;
    }
}

// ---------------------------------------------------------------------------
// Phase 2: flash attention.
// ---------------------------------------------------------------------------
__global__ __launch_bounds__(P2_WAVES * 32) void flash_attn_kernel(
    const half_t* __restrict__ Qh, const half_t* __restrict__ Kh,
    const half_t* __restrict__ Vt, float* __restrict__ out)
{
    __shared__ half_t sK[32 * 64];            // [key][d]     4 KB
    __shared__ half_t sV[64 * 32];            // [d][key]     4 KB
    __shared__ half_t sP[P2_WAVES][16 * 32];  // per-wave P   4 KB

    const int lane    = threadIdx.x & 31;
    const int wave    = threadIdx.x >> 5;
    const int halfsel = lane >> 4;
    const int l16     = lane & 15;
    const int qrow0   = blockIdx.x * (P2_WAVES * 16) + wave * 16;

    // LDS byte offsets for async staging (generic addr low 32 bits == LDS off)
    const unsigned ldsK0 = (unsigned)(size_t)(&sK[0]);
    const unsigned ldsV0 = (unsigned)(size_t)(&sV[0]);

    // Q A-fragments for k=0..31 and 32..63, resident in VGPRs.
    v16h qa[2];
#pragma unroll
    for (int s = 0; s < 2; ++s) {
        const half_t* qp = Qh + (size_t)(qrow0 + l16) * DOUT + s * 32 + 16 * halfsel;
        qa[s] = combine(*(const v8h*)qp, *(const v8h*)(qp + 8));
    }

    v8f acc[4];
#pragma unroll
    for (int jn = 0; jn < 4; ++jn) acc[jn] = zero_v8f();

    float mrow[8], lrow[8];
#pragma unroll
    for (int v = 0; v < 8; ++v) { mrow[v] = -1e30f; lrow[v] = 0.0f; }

    // softmax in log2 domain: scores * (1/sqrt(64)) * log2(e)
    const float scale = 0.125f * 1.44269504088896341f;

    for (int kb = 0; kb < SEQ; kb += 32) {
        // --- async staging of K/V tiles into LDS (ASYNCcnt path) ---
#pragma unroll
        for (int i = threadIdx.x; i < 256; i += P2_WAVES * 32) {
            const half_t* srcK = Kh + (size_t)kb * DOUT + i * 8;
            const unsigned dstK = ldsK0 + (unsigned)i * 16u;
            asm volatile("global_load_async_to_lds_b128 %0, %1, off"
                         :: "v"(dstK), "v"(srcK) : "memory");
            const int d = i >> 2, c0 = (i & 3) * 8;
            const half_t* srcV = Vt + (size_t)d * SEQ + kb + c0;
            const unsigned dstV = ldsV0 + (unsigned)i * 16u;
            asm volatile("global_load_async_to_lds_b128 %0, %1, off"
                         :: "v"(dstV), "v"(srcV) : "memory");
        }
        asm volatile("s_wait_asynccnt 0x0" ::: "memory");
        __syncthreads();

        // --- scores S[16x32] = Q(16x64) @ K^T(64x32): 2 n-frags x 2 k-steps
        v8f sc[2];
#pragma unroll
        for (int f = 0; f < 2; ++f) {
            v8f c = zero_v8f();
#pragma unroll
            for (int s = 0; s < 2; ++s) {
                const half_t* kp = sK + (f * 16 + l16) * DOUT + s * 32 + 16 * halfsel;
                const v16h b = combine(*(const v8h*)kp, *(const v8h*)(kp + 8));
                c = wmma_f16(qa[s], b, c);
            }
            sc[f] = c;
        }

        // --- online softmax; row stats replicated over 16 lanes per half
        float mx[8];
#pragma unroll
        for (int v = 0; v < 8; ++v) {
            sc[0][v] *= scale;
            sc[1][v] *= scale;
            mx[v] = fmaxf(sc[0][v], sc[1][v]);
        }
#pragma unroll
        for (int mask = 1; mask <= 8; mask <<= 1)
#pragma unroll
            for (int v = 0; v < 8; ++v)
                mx[v] = fmaxf(mx[v], __shfl_xor(mx[v], mask, 32));

        float ps[8], resc[8];
#pragma unroll
        for (int v = 0; v < 8; ++v) {
            const float mnew = fmaxf(mrow[v], mx[v]);
            resc[v] = exp2f(mrow[v] - mnew);
            const float p0 = exp2f(sc[0][v] - mnew);
            const float p1 = exp2f(sc[1][v] - mnew);
            mrow[v] = mnew;
            ps[v] = p0 + p1;
            sP[wave][(8 * halfsel + v) * 32 + l16]      = (half_t)p0;
            sP[wave][(8 * halfsel + v) * 32 + 16 + l16] = (half_t)p1;
#pragma unroll
            for (int jn = 0; jn < 4; ++jn) acc[jn][v] *= resc[v];
        }
#pragma unroll
        for (int mask = 1; mask <= 8; mask <<= 1)
#pragma unroll
            for (int v = 0; v < 8; ++v)
                ps[v] += __shfl_xor(ps[v], mask, 32);
#pragma unroll
        for (int v = 0; v < 8; ++v) lrow[v] = lrow[v] * resc[v] + ps[v];

        // --- P as A-fragment (16x32), V tile as B: acc += P @ V
        const half_t* pp = &sP[wave][l16 * 32 + 16 * halfsel];
        const v16h pa = combine(*(const v8h*)pp, *(const v8h*)(pp + 8));
#pragma unroll
        for (int jn = 0; jn < 4; ++jn) {
            const half_t* vp = sV + (jn * 16 + l16) * 32 + 16 * halfsel;
            const v16h b = combine(*(const v8h*)vp, *(const v8h*)(vp + 8));
            acc[jn] = wmma_f16(pa, b, acc[jn]);
        }
        __syncthreads();  // protect sK/sV before next tile's async writes
    }

    // --- epilogue: O / l
#pragma unroll
    for (int jn = 0; jn < 4; ++jn)
#pragma unroll
        for (int v = 0; v < 8; ++v) {
            const int r = qrow0 + 8 * halfsel + v;
            const int c = jn * 16 + l16;
            out[(size_t)r * DOUT + c] = acc[jn][v] / lrow[v];
        }
}

// ---------------------------------------------------------------------------
extern "C" void kernel_launch(void* const* d_in, const int* in_sizes, int n_in,
                              void* d_out, int out_size, void* d_ws, size_t ws_size,
                              hipStream_t stream)
{
    (void)in_sizes; (void)n_in; (void)out_size; (void)ws_size;

    const float* x  = (const float*)d_in[0];
    const float* Wq = (const float*)d_in[1];
    const float* Wk = (const float*)d_in[2];
    const float* Wv = (const float*)d_in[3];
    float* out = (float*)d_out;

    half_t* Qh = (half_t*)d_ws;                 // [SEQ][64] f16
    half_t* Kh = Qh + (size_t)SEQ * DOUT;       // [SEQ][64] f16
    half_t* Vt = Kh + (size_t)SEQ * DOUT;       // [64][SEQ] f16 (transposed)
    half_t* Wt = Vt + (size_t)SEQ * DOUT;       // [3][64][512] f16 (transposed)

    wcvt_kernel<<<(3 * DOUT * DIN) / 256, 256, 0, stream>>>(Wq, Wk, Wv, Wt);
    qkv_proj_kernel<<<SEQ / (P1_WAVES * 16), P1_WAVES * 32, 0, stream>>>(
        x, Wt, Qh, Kh, Vt);
    flash_attn_kernel<<<SEQ / (P2_WAVES * 16), P2_WAVES * 32, 0, stream>>>(
        Qh, Kh, Vt, out);
}
